// OpenAIMoeAttention_34136400068635
// MI455X (gfx1250) — matrix operand order repaired and checked by reading, not compiled, not run on hardware
//
#include <hip/hip_runtime.h>
#include <hip/hip_bf16.h>

// ---------------------------------------------------------------------------
// MI455X (gfx1250, wave32) implementation.
// bf16 WMMA (v_wmma_f32_16x16x32_bf16) for both projections and the
// flash-attention core; fp32 accumulation; weights converted to bf16 +
// transposed once (fit in 192MB L2). GEMM A-tiles staged in LDS through the
// gfx1250 async-copy path (global_load_async_to_lds_b128 + s_wait_asynccnt,
// via inline asm per the CDNA5 ISA since the builtin's pointer typing is
// toolchain-specific), double-buffered so the copy overlaps the WMMA stream.
// Sliding-window flash attention folds the sink into softmax init
// (m0 = log(sink), l0 = 1).
// ---------------------------------------------------------------------------

typedef __attribute__((ext_vector_type(16))) __bf16 v16bf;
typedef __attribute__((ext_vector_type(8)))  __bf16 v8bf;
typedef __attribute__((ext_vector_type(8)))  float  v8f;

#define CB   2
#define CS   1024
#define CH   2880
#define CNH  64
#define CNKV 8
#define CHD  64
#define CQS  4096      // NH*HD
#define CKVS 512       // NKV*HD
#define CQKVN 5120     // QS + 2*KVS
#define CTOK (CB*CS)   // 2048
#define CWIN 127
#define CSCALE 0.125f  // 64^-0.5
#define CLOGTHETA 11.9183893f // log(150000)

static __device__ inline v16bf join8(v8bf lo, v8bf hi) {
  union { v16bf w; v8bf h[2]; } u;
  u.h[0] = lo; u.h[1] = hi;
  return u.w;
}

static __device__ inline v8f wmma_bf16(v16bf a, v16bf b, v8f c) {
  // D = A(16x32 bf16) * B(32x16 bf16) + C(16x16 f32)
  return __builtin_amdgcn_wmma_f32_16x16x32_bf16(
      false, a, false, b, (short)0, c, false, false);
}

// Async 16-byte global -> LDS copy (ASYNCcnt-tracked, ISA 15.18.3 op 98).
// vdst = 32-bit LDS byte address (low 32 bits of generic LDS pointer),
// vaddr = 64-bit global address. Inline asm avoids the builtin's
// toolchain-specific pointer address-space typing.
static __device__ inline void async_copy16(const __bf16* g, __bf16* l) {
  asm volatile("global_load_async_to_lds_b128 %0, %1, off"
               :
               : "v"((unsigned)(size_t)l),
                 "v"((unsigned long long)(size_t)g)
               : "memory");
}

static __device__ inline void wait_async0() {
#if __has_builtin(__builtin_amdgcn_s_wait_asynccnt)
  __builtin_amdgcn_s_wait_asynccnt(0);
#else
  asm volatile("s_wait_asynccnt 0" ::: "memory");
#endif
}

// ---------------------------------------------------------------------------
// fp32 -> bf16 elementwise convert
// ---------------------------------------------------------------------------
__global__ void k_cvt(const float* __restrict__ in, __bf16* __restrict__ out, int n) {
  int i = blockIdx.x * blockDim.x + threadIdx.x;
  if (i < n) out[i] = (__bf16)in[i];
}

// ---------------------------------------------------------------------------
// fp32 [K x N] row-major  ->  bf16 BT[Npad x K] (transposed, zero-padded rows)
// LDS-tiled 32x32 transpose, block (32,8)
// ---------------------------------------------------------------------------
__global__ __launch_bounds__(256)
void k_transpose(const float* __restrict__ in, __bf16* __restrict__ out,
                 int K, int N, int Npad) {
  __shared__ float tile[32][33];
  const int bx = blockIdx.x * 32;  // n
  const int by = blockIdx.y * 32;  // k
  const int tx = threadIdx.x, ty = threadIdx.y;
#pragma unroll
  for (int j = 0; j < 4; ++j) {
    int k = by + ty + j * 8, n = bx + tx;
    tile[ty + j * 8][tx] = (n < N && k < K) ? in[(size_t)k * N + n] : 0.f;
  }
  __syncthreads();
#pragma unroll
  for (int j = 0; j < 4; ++j) {
    int n = bx + ty + j * 8, k = by + tx;
    if (n < Npad && k < K)
      out[(size_t)n * K + k] = (__bf16)tile[tx][ty + j * 8];
  }
}

// ---------------------------------------------------------------------------
// C[M x N] f32 = A[M x K] bf16 (row-major) @ BT[Npad x K] bf16 (B transposed)
// 256 threads = 8 waves; block tile 128(M) x 256(N); wave tile 64x64
// => 16 v_wmma_f32_16x16x32_bf16 per 32-deep K step per wave.
// A block-tile (128x32) staged in LDS via async copy, double-buffered:
// copy of step k+32 overlaps WMMAs of step k; one s_wait_asynccnt + barrier
// per step. 80B padded LDS rows -> conflict-free ds reads (20*row mod 64
// distinct over 16 lanes), 16B-aligned chunks.
// A-fragment (contiguous k, same order as B): lane<16 holds K {0..15},
// lane>=16 holds K {16..31}; row = lane%16.
// B-fragment: lane<16 col=lane K=0..15; lane>=16 K=16..31 (contiguous from BT).
// ---------------------------------------------------------------------------
__global__ __launch_bounds__(256)
void k_gemm(const __bf16* __restrict__ A, const __bf16* __restrict__ BT,
            float* __restrict__ C, int M, int K, int Nstore, int ldc) {
  __shared__ __align__(16) __bf16 Abuf[2][128][40];   // 2 x 10KB, padded rows

  const int tid  = threadIdx.x;
  const int wave = tid >> 5, lane = tid & 31;
  const int half = lane >> 4, ln = lane & 15;
  const int waveM = wave & 1, waveN = wave >> 1;
  const int rowBlk  = blockIdx.y * 128;
  const int rowBase = rowBlk + waveM * 64;
  const int colBase = blockIdx.x * 256 + waveN * 64;
  const int o0 = half * 16, o1 = half * 16 + 8;

  v8f acc[4][4];
#pragma unroll
  for (int mf = 0; mf < 4; ++mf)
#pragma unroll
    for (int nf = 0; nf < 4; ++nf)
#pragma unroll
      for (int r = 0; r < 8; ++r) acc[mf][nf][r] = 0.f;

  const __bf16* Brow[4];
#pragma unroll
  for (int nf = 0; nf < 4; ++nf)
    Brow[nf] = BT + (size_t)(colBase + nf * 16 + ln) * K + half * 16;

  // 128x32 bf16 A tile = 512 16-byte chunks; 2 per thread
  auto stageA = [&](int buf, int kk) {
#pragma unroll
    for (int j = 0; j < 2; ++j) {
      int e = tid * 2 + j;               // 0..511
      int r = e >> 2, c = (e & 3) * 8;   // row 0..127, col chunk {0,8,16,24}
      const __bf16* g = A + (size_t)(rowBlk + r) * K + kk + c;
      async_copy16(g, &Abuf[buf][r][c]);
    }
  };

  stageA(0, 0);
  wait_async0();
  __syncthreads();

  const int rowLoc = waveM * 64 + ln;
  for (int kk = 0; kk < K; kk += 32) {
    const int buf = (kk >> 5) & 1;
    if (kk + 32 < K) stageA(buf ^ 1, kk + 32);   // overlap copy with WMMAs

    v16bf a[4];
#pragma unroll
    for (int mf = 0; mf < 4; ++mf)
      a[mf] = join8(*(const v8bf*)(&Abuf[buf][rowLoc + mf * 16][o0]),
                    *(const v8bf*)(&Abuf[buf][rowLoc + mf * 16][o1]));
#pragma unroll
    for (int nf = 0; nf < 4; ++nf) {
      v16bf bfrag = *(const v16bf*)(Brow[nf] + kk);
#pragma unroll
      for (int mf = 0; mf < 4; ++mf)
        acc[mf][nf] = wmma_bf16(a[mf], bfrag, acc[mf][nf]);
    }

    if (kk + 256 < K) {  // global_prefetch_b8: warm future B K-tile into L2
      __builtin_prefetch(Brow[0] + kk + 256, 0, 3);
      __builtin_prefetch(Brow[2] + kk + 256, 0, 3);
    }
    if (kk + 32 < K) {
      wait_async0();       // __syncthreads does not cover ASYNCcnt
      __syncthreads();
    }
  }

  // C/D layout: VGPR r -> M = r + 8*(lane>=16); N = lane%16
#pragma unroll
  for (int mf = 0; mf < 4; ++mf)
#pragma unroll
    for (int nf = 0; nf < 4; ++nf)
#pragma unroll
      for (int r = 0; r < 8; ++r) {
        int row = rowBase + mf * 16 + r + 8 * half;
        int col = colBase + nf * 16 + ln;
        if (col < Nstore) C[(size_t)row * ldc + col] = acc[mf][nf][r];
      }
}

// ---------------------------------------------------------------------------
// RoPE (neox): qkv f32 [2048 x 5120] -> roped q bf16 [2048 x 4096],
// roped k bf16 [2048 x 512], v bf16 [2048 x 512]. One block per token.
// ---------------------------------------------------------------------------
__global__ __launch_bounds__(256)
void k_rope(const float* __restrict__ qkv, const int* __restrict__ pos,
            __bf16* __restrict__ q, __bf16* __restrict__ k, __bf16* __restrict__ v) {
  const int tok = blockIdx.x;
  const int t = threadIdx.x;
  const float p = (float)pos[tok];
  const float* src = qkv + (size_t)tok * CQKVN;

  // Q heads: 64 heads x 32 pairs = 2048 pairs, 8 per thread
#pragma unroll
  for (int j = 0; j < 8; ++j) {
    int pr = t + j * 256;
    int hh = pr >> 5, d = pr & 31;
    float inv = __expf((float)d * (-CLOGTHETA / 32.f));
    float s, c;
    __sincosf(p * inv, &s, &c);
    float x1 = src[hh * 64 + d], x2 = src[hh * 64 + d + 32];
    q[(size_t)tok * CQS + hh * 64 + d]      = (__bf16)(x1 * c - x2 * s);
    q[(size_t)tok * CQS + hh * 64 + d + 32] = (__bf16)(x2 * c + x1 * s);
  }
  // K heads: 8 heads x 32 pairs = 256 pairs, 1 per thread
  {
    int hh = t >> 5, d = t & 31;
    float inv = __expf((float)d * (-CLOGTHETA / 32.f));
    float s, c;
    __sincosf(p * inv, &s, &c);
    float x1 = src[CQS + hh * 64 + d], x2 = src[CQS + hh * 64 + d + 32];
    k[(size_t)tok * CKVS + hh * 64 + d]      = (__bf16)(x1 * c - x2 * s);
    k[(size_t)tok * CKVS + hh * 64 + d + 32] = (__bf16)(x2 * c + x1 * s);
  }
  // V passthrough convert: 512 elems, 2 per thread
#pragma unroll
  for (int j = 0; j < 2; ++j) {
    int e = t + j * 256;
    v[(size_t)tok * CKVS + e] = (__bf16)src[CQS + CKVS + e];
  }
}

// ---------------------------------------------------------------------------
// Sliding-window flash attention with sinks, all-WMMA.
// grid (S/64, NH, B), block 128 (4 waves); wave handles 16 query rows.
// Score: Q(16x64) @ K^T -> two 16x16 tiles per 32-key step (4 WMMAs).
// Online softmax: running max init = log(sink), running sum init = 1
// (exactly reproduces the sink column of the reference softmax).
// P routed through LDS (80B row stride: conflict-free, 16B aligned);
// V staged transposed in LDS so P@V B-fragments are contiguous ds loads.
// ---------------------------------------------------------------------------
__global__ __launch_bounds__(128)
void k_attn(const __bf16* __restrict__ Q, const __bf16* __restrict__ Kc,
            const __bf16* __restrict__ Vc, const float* __restrict__ sinks,
            __bf16* __restrict__ Ao) {
  __shared__ __align__(16) __bf16 VT[64][40];   // V^T tile: [hd][key_local]
  __shared__ __align__(16) __bf16 P[4][16][40]; // per-wave P tile: [m][key_local]

  const int tid  = threadIdx.x;
  const int wave = tid >> 5, lane = tid & 31;
  const int half = lane >> 4, ln = lane & 15;
  const int t0 = blockIdx.x * 64;
  const int h  = blockIdx.y;
  const int b  = blockIdx.z;
  const int kvh = h >> 3;             // NH / NKV = 8
  const int qrow0 = t0 + wave * 16;
  const int o0 = half * 16, o1 = half * 16 + 8;

  // Q fragments, 2 chunks over HD=64 (A-fragment layout, row = lane%16)
  const __bf16* qbase = Q + (size_t)(b * CS + qrow0 + ln) * CQS + h * CHD;
  v16bf qa[2];
#pragma unroll
  for (int c = 0; c < 2; ++c)
    qa[c] = join8(*(const v8bf*)(qbase + c * 32 + o0),
                  *(const v8bf*)(qbase + c * 32 + o1));

  float m[8], l[8];
  v8f Oa[4];
  const float lsink = __logf(sinks[h]);
#pragma unroll
  for (int r = 0; r < 8; ++r) { m[r] = lsink; l[r] = 1.f; }
#pragma unroll
  for (int nf = 0; nf < 4; ++nf)
#pragma unroll
    for (int r = 0; r < 8; ++r) Oa[nf][r] = 0.f;

  int ks = t0 - CWIN;
  if (ks < 0) ks = 0;
  ks &= ~31;
  const int ke = t0 + 63;

  for (int kt = ks; kt <= ke; kt += 32) {
    // ---- scores: two 16-key subtiles ----
    v8f s[2];
#pragma unroll
    for (int sub = 0; sub < 2; ++sub) {
      const __bf16* krow =
          Kc + (size_t)(b * CS + kt + sub * 16 + ln) * CKVS + kvh * CHD;
      v16bf kb0 = *(const v16bf*)(krow + half * 16);        // hd 0..31
      v16bf kb1 = *(const v16bf*)(krow + 32 + half * 16);   // hd 32..63
      v8f z;
#pragma unroll
      for (int r = 0; r < 8; ++r) z[r] = 0.f;
      z = wmma_bf16(qa[0], kb0, z);
      s[sub] = wmma_bf16(qa[1], kb1, z);
    }

    // ---- online softmax (per row, cross-lane reduce within 16-group) ----
#pragma unroll
    for (int r = 0; r < 8; ++r) {
      const int row = qrow0 + r + 8 * half;
      const int c0 = kt + ln, c1 = kt + 16 + ln;
      float a0 = (c0 <= row && c0 >= row - CWIN) ? s[0][r] * CSCALE : -1e30f;
      float a1 = (c1 <= row && c1 >= row - CWIN) ? s[1][r] * CSCALE : -1e30f;
      float tmax = fmaxf(a0, a1);
#pragma unroll
      for (int off = 8; off; off >>= 1)
        tmax = fmaxf(tmax, __shfl_xor(tmax, off, 32));
      const float mn = fmaxf(m[r], tmax);
      const float alpha = __expf(m[r] - mn);
      const float p0 = __expf(a0 - mn), p1 = __expf(a1 - mn);
      float rs = p0 + p1;
#pragma unroll
      for (int off = 8; off; off >>= 1) rs += __shfl_xor(rs, off, 32);
      l[r] = l[r] * alpha + rs;
      m[r] = mn;
#pragma unroll
      for (int nf = 0; nf < 4; ++nf) Oa[nf][r] *= alpha;
      P[wave][r + 8 * half][ln]      = (__bf16)p0;
      P[wave][r + 8 * half][16 + ln] = (__bf16)p1;
    }

    // ---- stage V^T (shared by all 4 waves; loop count is block-uniform) ----
    __syncthreads();
    for (int e = tid; e < 32 * 64; e += 128) {
      int kk = e >> 6, hd = e & 63;
      VT[hd][kk] = Vc[(size_t)(b * CS + kt + kk) * CKVS + kvh * CHD + hd];
    }
    __syncthreads();

    // ---- P @ V: A from LDS P, B from LDS V^T ----
    v16bf pa = join8(*(const v8bf*)(&P[wave][ln][o0]),
                     *(const v8bf*)(&P[wave][ln][o1]));
#pragma unroll
    for (int nf = 0; nf < 4; ++nf) {
      const __bf16* vr = &VT[nf * 16 + ln][half * 16];
      v16bf vb = join8(*(const v8bf*)(vr), *(const v8bf*)(vr + 8));
      Oa[nf] = wmma_bf16(pa, vb, Oa[nf]);
    }
  }

  // ---- epilogue: O / l -> bf16 ----
  float inv[8];
#pragma unroll
  for (int r = 0; r < 8; ++r) inv[r] = 1.f / l[r];
#pragma unroll
  for (int nf = 0; nf < 4; ++nf)
#pragma unroll
    for (int r = 0; r < 8; ++r) {
      int tok = b * CS + qrow0 + r + 8 * half;
      Ao[(size_t)tok * CQS + h * CHD + nf * 16 + ln] =
          (__bf16)(Oa[nf][r] * inv[r]);
    }
}

// ---------------------------------------------------------------------------
extern "C" void kernel_launch(void* const* d_in, const int* in_sizes, int n_in,
                              void* d_out, int out_size, void* d_ws, size_t ws_size,
                              hipStream_t stream) {
  (void)in_sizes; (void)n_in; (void)out_size; (void)ws_size;
  const int*   positions = (const int*)d_in[0];
  const float* hidden    = (const float*)d_in[1];
  const float* wqkv      = (const float*)d_in[2];
  const float* wo        = (const float*)d_in[3];
  const float* sinks     = (const float*)d_in[4];
  float* out = (float*)d_out;

  char* ws = (char*)d_ws;
  auto alloc = [&](size_t bytes) -> char* {
    char* p = ws;
    ws += (bytes + 255) & ~(size_t)255;
    return p;
  };
  __bf16* hsb   = (__bf16*)alloc((size_t)CTOK * CH * 2);        // hidden bf16
  __bf16* wqkvT = (__bf16*)alloc((size_t)CQKVN * CH * 2);       // w_qkv^T bf16
  __bf16* woT   = (__bf16*)alloc((size_t)3072 * CQS * 2);       // w_o^T bf16 (padded N)
  float*  qkv   = (float*) alloc((size_t)CTOK * CQKVN * 4);     // qkv f32
  __bf16* qb    = (__bf16*)alloc((size_t)CTOK * CQS * 2);       // roped q
  __bf16* kb    = (__bf16*)alloc((size_t)CTOK * CKVS * 2);      // roped k
  __bf16* vb    = (__bf16*)alloc((size_t)CTOK * CKVS * 2);      // v
  __bf16* ab    = (__bf16*)alloc((size_t)CTOK * CQS * 2);       // attn out

  // 1) converts / transposes (weights land in L2 as bf16)
  k_cvt<<<(CTOK * CH + 255) / 256, 256, 0, stream>>>(hidden, hsb, CTOK * CH);
  k_transpose<<<dim3(CQKVN / 32, CH / 32), dim3(32, 8), 0, stream>>>(
      wqkv, wqkvT, CH, CQKVN, CQKVN);
  k_transpose<<<dim3(3072 / 32, CQS / 32), dim3(32, 8), 0, stream>>>(
      wo, woT, CQS, CH, 3072);

  // 2) QKV projection: [2048x2880] @ [2880x5120]
  k_gemm<<<dim3(CQKVN / 256, CTOK / 128), 256, 0, stream>>>(
      hsb, wqkvT, qkv, CTOK, CH, CQKVN, CQKVN);

  // 3) RoPE + bf16 repack
  k_rope<<<CTOK, 256, 0, stream>>>(qkv, positions, qb, kb, vb);

  // 4) sliding-window flash attention with sinks
  k_attn<<<dim3(CS / 64, CNH, CB), 128, 0, stream>>>(qb, kb, vb, sinks, ab);

  // 5) output projection: [2048x4096] @ [4096x2880]
  k_gemm<<<dim3(3072 / 256, CTOK / 128), 256, 0, stream>>>(
      ab, woT, out, CTOK, CQS, 2880, 2880);
}
